// QASoftMaxAttention_20366734917932
// MI455X (gfx1250) — matrix operand order repaired and checked by reading, not compiled, not run on hardware
//
#include <hip/hip_runtime.h>
#include <hip/hip_bf16.h>

#define HID   1024
#define SEQ   2048
#define BATCH 8
#define NKB   (SEQ / 32)
#define SCALE 0.03125f   // 1/sqrt(1024)

typedef __attribute__((ext_vector_type(16))) _Float16 v16h;
typedef __attribute__((ext_vector_type(8)))  _Float16 v8h;
typedef __attribute__((ext_vector_type(4)))  _Float16 v4h;
typedef __attribute__((ext_vector_type(8)))  float    v8f;

// ---------------------------------------------------------------------------
// Pre-pass 1: QK[b,s,h] = f16(seq * mask)   (same layout as seq)
// ---------------------------------------------------------------------------
__global__ void cvt_qk_kernel(const float* __restrict__ seq,
                              const float* __restrict__ mask,
                              _Float16* __restrict__ qk) {
  size_t base = ((size_t)blockIdx.x * 256 + threadIdx.x) * 4;
  float m = mask[base >> 10];                 // base / HID -> (b*SEQ + s)
  float4 v = *(const float4*)(seq + base);
  v4h o;
  o[0] = (_Float16)(v.x * m);
  o[1] = (_Float16)(v.y * m);
  o[2] = (_Float16)(v.z * m);
  o[3] = (_Float16)(v.w * m);
  *(v4h*)(qk + base) = o;
}

// ---------------------------------------------------------------------------
// Pre-pass 2: Vt[b,h,s] = f16(seq[b,s,h])   (transposed, LDS-tiled)
// ---------------------------------------------------------------------------
__global__ void cvt_vt_kernel(const float* __restrict__ seq,
                              _Float16* __restrict__ vt) {
  __shared__ _Float16 tile[32][33];
  const int b  = blockIdx.z;
  const int s0 = blockIdx.x * 32;
  const int h0 = blockIdx.y * 32;
  const int tx = threadIdx.x;        // 0..31
  const int ty = threadIdx.y;        // 0..7
#pragma unroll
  for (int r = 0; r < 4; ++r) {
    int s = s0 + ty + r * 8;
    tile[ty + r * 8][tx] = (_Float16)seq[((size_t)b * SEQ + s) * HID + h0 + tx];
  }
  __syncthreads();
#pragma unroll
  for (int r = 0; r < 4; ++r) {
    int h = h0 + ty + r * 8;
    vt[((size_t)b * HID + h) * SEQ + s0 + tx] = tile[tx][ty + r * 8];
  }
}

// ---------------------------------------------------------------------------
// Flash attention: one workgroup = 16 queries, 8 waves (wave32).
//  wave w: QK^T partial over hidden slice (w>>1)*256, key sub-tile (w&1);
//          P*V over output-dim slice w*128 (8 accumulator tiles of 16x16 f32).
//  B-fragments are software-pipelined one key-block ahead; score partials go
//  to a conflict-free 4-way LDS buffer summed during softmax (no atomics).
// ---------------------------------------------------------------------------
__global__ void __launch_bounds__(256)
flash_attn_kernel(const _Float16* __restrict__ qk,
                  const _Float16* __restrict__ vt,
                  float* __restrict__ out) {
  __shared__ float st4[4][16][32];                               // score partials
  __shared__ __attribute__((aligned(64))) _Float16 Pt[16][32];   // probs (f16)
  __shared__ float m_s[16], l_s[16], cf_s[16];

  const int b     = blockIdx.y;
  const int q0    = blockIdx.x * 16;
  const int tid   = threadIdx.x;
  const int lane  = tid & 31;
  const int wv    = tid >> 5;         // wave 0..7
  const int ntile = wv & 1;           // key sub-tile within 32-key block
  const int hsl   = wv >> 1;          // hidden-slice index 0..3
  const int h0    = hsl * 256;        // hidden slice for QK^T
  const int d0    = wv * 128;         // output dim slice for P*V
  const int lo16  = lane & 15;
  const int half  = lane >> 4;        // 0 | 1
  const int kbA   = half * 16;        // A-fragment K base (contiguous, matches B)
  const int kbB   = half * 16;        // B-fragment K base (contiguous, matches A)
  const int rbase = half * 8;         // C/D row base

  if (tid < 16) { m_s[tid] = -1e30f; l_s[tid] = 0.f; }

  // ---- preload Q fragments (A matrices) for this wave's hidden slice ----
  const _Float16* qrow = qk + ((size_t)b * SEQ + q0 + lo16) * HID;
  v16h qfrag[8];
#pragma unroll
  for (int kk = 0; kk < 8; ++kk) {
    const int kcol = h0 + kk * 32;
    v8h alo = *(const v8h*)(qrow + kcol + kbA);
    v8h ahi = *(const v8h*)(qrow + kcol + kbA + 8);
    qfrag[kk] = __builtin_shufflevector(alo, ahi,
                  0, 1, 2, 3, 4, 5, 6, 7, 8, 9, 10, 11, 12, 13, 14, 15);
  }

  v8f zero = {};
  v8f acc[8];
#pragma unroll
  for (int t = 0; t < 8; ++t) acc[t] = zero;

  // K-side B-fragment base (advances 32 keys per block)
  const _Float16* krow0 = qk + ((size_t)b * SEQ + ntile * 16 + lo16) * HID
                             + h0 + kbB;
  // V-side B-fragment base (advances 32 along the key axis per block)
  const _Float16* vcol0 = vt + ((size_t)b * HID + d0 + lo16) * SEQ + kbB;

  // ---- software pipeline: preload B-fragments for key block 0 ----
  v16h bfr[8];
#pragma unroll
  for (int kk = 0; kk < 8; ++kk)
    bfr[kk] = *(const v16h*)(krow0 + kk * 32);

  for (int kb = 0; kb < NKB; ++kb) {
    __syncthreads();                     // st4 / Pt consumers of prev iter done

    // ---------------- QK^T partial (8 WMMAs over 256 hidden dims) ---------
    v8f c = zero;
#pragma unroll
    for (int kk = 0; kk < 8; ++kk)
      c = __builtin_amdgcn_wmma_f32_16x16x32_f16(
              false, qfrag[kk], false, bfr[kk], (short)0, c, false, false);

#pragma unroll
    for (int j = 0; j < 8; ++j)          // conflict-free partial store
      st4[hsl][rbase + j][ntile * 16 + lo16] = c[j];

    // ---- prefetch next key block's B-fragments (overlap softmax + P*V) ---
    if (kb + 1 < NKB) {
      const _Float16* krow = krow0 + (size_t)(kb + 1) * 32 * HID;
      __builtin_prefetch(krow + 32 * HID, 0, 1);   // two blocks ahead -> L2
#pragma unroll
      for (int kk = 0; kk < 8; ++kk)
        bfr[kk] = *(const v16h*)(krow + kk * 32);
    }

    // ---- V fragments: depend only on kb, overlap with softmax ------------
    const _Float16* vcol = vcol0 + kb * 32;
    v16h vfr[8];
#pragma unroll
    for (int t = 0; t < 8; ++t)
      vfr[t] = *(const v16h*)(vcol + (size_t)t * 16 * SEQ); // contiguous 32B

    __syncthreads();

    // ---------------- online softmax: 2 rows per wave ---------------------
#pragma unroll
    for (int rr = 0; rr < 2; ++rr) {
      const int r = wv * 2 + rr;
      float s = (st4[0][r][lane] + st4[1][r][lane] +
                 st4[2][r][lane] + st4[3][r][lane]) * SCALE;
      float mx = s;
      for (int off = 16; off; off >>= 1) mx = fmaxf(mx, __shfl_xor(mx, off, 32));
      float mo = m_s[r];
      float mn = fmaxf(mo, mx);
      float p  = __expf(s - mn);
      float ps = p;
      for (int off = 16; off; off >>= 1) ps += __shfl_xor(ps, off, 32);
      Pt[r][lane] = (_Float16)p;
      if (lane == 0) {
        float cr = __expf(mo - mn);
        cf_s[r]  = cr;
        m_s[r]   = mn;
        l_s[r]   = l_s[r] * cr + ps;
      }
    }
    __syncthreads();

    // ---------------- rescale accumulators + P*V (8 WMMAs) ----------------
    float cfl[8];
#pragma unroll
    for (int j = 0; j < 8; ++j) cfl[j] = cf_s[rbase + j];

    v8h plo = *(const v8h*)(&Pt[lo16][kbA]);
    v8h phi = *(const v8h*)(&Pt[lo16][kbA + 8]);
    v16h pfrag = __builtin_shufflevector(plo, phi,
                   0, 1, 2, 3, 4, 5, 6, 7, 8, 9, 10, 11, 12, 13, 14, 15);

#pragma unroll
    for (int t = 0; t < 8; ++t) {
#pragma unroll
      for (int j = 0; j < 8; ++j) acc[t][j] *= cfl[j];
      acc[t] = __builtin_amdgcn_wmma_f32_16x16x32_f16(
                   false, pfrag, false, vfr[t], (short)0, acc[t], false, false);
    }
  }

  __syncthreads();
  float invl[8];
#pragma unroll
  for (int j = 0; j < 8; ++j) invl[j] = 1.0f / l_s[rbase + j];
#pragma unroll
  for (int t = 0; t < 8; ++t) {
#pragma unroll
    for (int j = 0; j < 8; ++j) {
      out[((size_t)b * SEQ + q0 + rbase + j) * HID + d0 + t * 16 + lo16] =
          acc[t][j] * invl[j];
    }
  }
}

// ---------------------------------------------------------------------------
extern "C" void kernel_launch(void* const* d_in, const int* in_sizes, int n_in,
                              void* d_out, int out_size, void* d_ws, size_t ws_size,
                              hipStream_t stream) {
  const float* seq  = (const float*)d_in[0];   // [8,2048,1024] f32
  const float* mask = (const float*)d_in[1];   // [8,2048] f32
  float* out = (float*)d_out;                  // [8,2048,1024] f32

  const size_t nelem = (size_t)BATCH * SEQ * HID;
  _Float16* qk = (_Float16*)d_ws;                      // 32 MB f16 masked Q/K
  _Float16* vt = (_Float16*)((char*)d_ws + nelem * 2); // 32 MB f16 V transposed

  cvt_qk_kernel<<<(unsigned)(nelem / (256 * 4)), 256, 0, stream>>>(seq, mask, qk);
  dim3 tg(SEQ / 32, HID / 32, BATCH);
  cvt_vt_kernel<<<tg, dim3(32, 8), 0, stream>>>(seq, vt);
  dim3 ag(SEQ / 16, BATCH);
  flash_attn_kernel<<<ag, 256, 0, stream>>>(qk, vt, out);
}
